// PolarQuantizer_59880434040869
// MI455X (gfx1250) — compile-verified, matrix operand order checked by reading, not run on hardware
//
#include <hip/hip_runtime.h>
#include <stdint.h>

#define DIM 128
#define TILE_M 16
#define WAVES_PER_BLOCK 8
#define BLOCK_THREADS (WAVES_PER_BLOCK * 32)
#define NUM_BLOCKS 512

typedef __attribute__((ext_vector_type(16))) __bf16 v16bf;
typedef __attribute__((ext_vector_type(8)))  float  v8f;

union Frag {
    unsigned int u[8];
    uint4        q[2];
    v16bf        v;
};

// fp32 -> bf16 with round-to-nearest-even, pure integer (no header deps)
__device__ __forceinline__ unsigned int bf16_rne(float f) {
    unsigned int u = __float_as_uint(f);
    return (u + 0x7FFFu + ((u >> 16) & 1u)) >> 16;
}
__device__ __forceinline__ float bf16_to_f32(unsigned int h) {
    return __uint_as_float(h << 16);
}

// WMMA 16-bit operand K-pair pattern (ISA 7.12.2): VGPR r, lane-half hf
__device__ __forceinline__ int klocal(int r, int hf) {
    return 2 * r + ((r >= 4) ? 8 : 0) + 8 * hf;
}

__global__ __launch_bounds__(BLOCK_THREADS, 1)
void polar_quant_wmma(const float* __restrict__ x,
                      const float* __restrict__ R,
                      float* __restrict__ codes,   // [N,128] as float
                      float* __restrict__ scales,  // [N]
                      int nTiles) {
    // B fragments: [t(8)][kc(4)][hi/lo(2)] x 256 dwords (32 lanes x 8 VGPRs) = 64 KB
    __shared__ unsigned int ldsB[16384];

    const int tid = threadIdx.x;

    // ---- Stage R as hi/lo bf16 B-fragments in WMMA register layout ----
    for (int i = 0; i < 16384 / BLOCK_THREADS; ++i) {
        int d    = tid + i * BLOCK_THREADS;   // [13:8]=frag [7:3]=lane [2:0]=r
        int r    = d & 7;
        int lane = (d >> 3) & 31;
        int f    = d >> 8;
        int hsel = f & 1;                     // 0 = hi, 1 = lo
        int kc   = (f >> 1) & 3;
        int t    = f >> 3;
        int hf   = lane >> 4;
        int n    = t * 16 + (lane & 15);
        int k    = kc * 32 + klocal(r, hf);
        float e0 = R[(size_t)k * DIM + n];
        float e1 = R[(size_t)(k + 1) * DIM + n];
        unsigned int h0 = bf16_rne(e0), h1 = bf16_rne(e1);
        unsigned int w;
        if (hsel == 0) {
            w = (h1 << 16) | h0;
        } else {
            unsigned int l0 = bf16_rne(e0 - bf16_to_f32(h0));
            unsigned int l1 = bf16_rne(e1 - bf16_to_f32(h1));
            w = (l1 << 16) | l0;
        }
        ldsB[d] = w;
    }
    __syncthreads();

    const int lane = tid & 31;
    const int wid  = tid >> 5;
    const int hf   = lane >> 4;    // lane-half: selects K sub-pattern / row half
    const int ln   = lane & 15;    // column within a 16-wide tile / A row
    const int gwave  = blockIdx.x * WAVES_PER_BLOCK + wid;
    const int nWaves = gridDim.x * WAVES_PER_BLOCK;

    for (int tile = gwave; tile < nTiles; tile += nWaves) {
        const int row0 = tile * TILE_M;
        const float* xrow = x + (size_t)(row0 + ln) * DIM;

        // Prefetch next tile's 8 KB of x (32 lanes x 256 B) -> global_prefetch_b8
        int nextTile = tile + nWaves;
        if (nextTile < nTiles) {
            __builtin_prefetch(x + (size_t)nextTile * TILE_M * DIM + lane * 64, 0, 1);
        }

        // ---- Load x rows and split into bf16 hi/lo A fragments ----
        Frag ah[4], al[4];
#pragma unroll
        for (int kc = 0; kc < 4; ++kc) {
            int k0 = kc * 32 + 8 * hf;       // run for VGPRs 0..3
            int k1 = k0 + 16;                // run for VGPRs 4..7
            float4 f0 = *(const float4*)(xrow + k0);
            float4 f1 = *(const float4*)(xrow + k0 + 4);
            float4 f2 = *(const float4*)(xrow + k1);
            float4 f3 = *(const float4*)(xrow + k1 + 4);
            float vals[16] = {f0.x, f0.y, f0.z, f0.w, f1.x, f1.y, f1.z, f1.w,
                              f2.x, f2.y, f2.z, f2.w, f3.x, f3.y, f3.z, f3.w};
#pragma unroll
            for (int r = 0; r < 8; ++r) {
                float e0 = vals[2 * r], e1 = vals[2 * r + 1];
                unsigned int h0 = bf16_rne(e0), h1 = bf16_rne(e1);
                unsigned int l0 = bf16_rne(e0 - bf16_to_f32(h0));
                unsigned int l1 = bf16_rne(e1 - bf16_to_f32(h1));
                ah[kc].u[r] = (h1 << 16) | h0;
                al[kc].u[r] = (l1 << 16) | l0;
            }
        }

        // ---- GEMM: 8 column tiles x 4 K-chunks x (hi*hi + hi*lo + lo*hi) ----
        v8f acc[8];
        const v8f vzero = {0.f, 0.f, 0.f, 0.f, 0.f, 0.f, 0.f, 0.f};
#pragma unroll
        for (int t = 0; t < 8; ++t) acc[t] = vzero;

#pragma unroll
        for (int t = 0; t < 8; ++t) {
#pragma unroll
            for (int kc = 0; kc < 4; ++kc) {
                Frag bh, bl;
                const uint4* ph =
                    (const uint4*)&ldsB[(((t * 4 + kc) * 2 + 0) << 8) + lane * 8];
                bh.q[0] = ph[0];
                bh.q[1] = ph[1];
                const uint4* pl =
                    (const uint4*)&ldsB[(((t * 4 + kc) * 2 + 1) << 8) + lane * 8];
                bl.q[0] = pl[0];
                bl.q[1] = pl[1];
                acc[t] = __builtin_amdgcn_wmma_f32_16x16x32_bf16(
                    false, ah[kc].v, false, bh.v, (short)0, acc[t], false, false);
                acc[t] = __builtin_amdgcn_wmma_f32_16x16x32_bf16(
                    false, ah[kc].v, false, bl.v, (short)0, acc[t], false, false);
                acc[t] = __builtin_amdgcn_wmma_f32_16x16x32_bf16(
                    false, al[kc].v, false, bh.v, (short)0, acc[t], false, false);
            }
        }

        // ---- Per-row max-abs reduction (across tiles, then across 16 lanes) ----
        float sc[8];
#pragma unroll
        for (int v = 0; v < 8; ++v) {
            float m = 0.f;
#pragma unroll
            for (int t = 0; t < 8; ++t) m = fmaxf(m, fabsf(acc[t][v]));
            m = fmaxf(m, __shfl_xor(m, 1, 16));
            m = fmaxf(m, __shfl_xor(m, 2, 16));
            m = fmaxf(m, __shfl_xor(m, 4, 16));
            m = fmaxf(m, __shfl_xor(m, 8, 16));
            sc[v] = fmaxf(m, 1e-8f);
        }

        // ---- Quantize: clip(round((v/s + 1) * 3.5), 0, 7) ----
#pragma unroll
        for (int v = 0; v < 8; ++v) {
            float inv = 3.5f / sc[v];
            int rowg = row0 + v + 8 * hf;
            float* orow = codes + (size_t)rowg * DIM + ln;
#pragma unroll
            for (int t = 0; t < 8; ++t) {
                float q = rintf(fmaf(acc[t][v], inv, 3.5f));
                q = fminf(fmaxf(q, 0.f), 7.f);
                orow[t * 16] = q;
            }
        }
        if (ln == 0) {
#pragma unroll
            for (int v = 0; v < 8; ++v) scales[row0 + v + 8 * hf] = sc[v];
        }
    }
}

extern "C" void kernel_launch(void* const* d_in, const int* in_sizes, int n_in,
                              void* d_out, int out_size, void* d_ws, size_t ws_size,
                              hipStream_t stream) {
    const float* x = (const float*)d_in[0];   // [N, 128] fp32
    const float* R = (const float*)d_in[1];   // [128, 128] fp32
    const int N = in_sizes[0] / DIM;
    float* codes  = (float*)d_out;            // N*128 code values (as float)
    float* scales = codes + (size_t)N * DIM;  // N scales
    const int nTiles = N / TILE_M;

    polar_quant_wmma<<<NUM_BLOCKS, BLOCK_THREADS, 0, stream>>>(x, R, codes, scales,
                                                               nTiles);
}
